// AttentionBlock_23776938951165
// MI455X (gfx1250) — compile-verified, matrix operand order checked by reading, not run on hardware
//
#include <hip/hip_runtime.h>

#define BB 4
#define SS 2048
#define DD 1024
#define HH 16
#define DHD 64
#define MROWS (BB * SS) // 8192

#ifndef __has_builtin
#define __has_builtin(x) 0
#endif
#if __has_builtin(__builtin_amdgcn_tensor_load_to_lds) && \
    __has_builtin(__builtin_amdgcn_s_wait_tensorcnt)
#define USE_TDM 1
#else
#define USE_TDM 0
#endif

typedef __attribute__((ext_vector_type(16))) __bf16 v16bf;
typedef __attribute__((ext_vector_type(8)))  float  v8f;

union FragBF { v16bf v; unsigned int u[8]; };

__device__ __forceinline__ unsigned short f32_to_bf16(float f) {
    unsigned int x = __float_as_uint(f);
    x += 0x7FFFu + ((x >> 16) & 1u);   // round-to-nearest-even
    return (unsigned short)(x >> 16);
}

__device__ __forceinline__ v8f v8f_zero() {
    v8f z;
#pragma unroll
    for (int i = 0; i < 8; ++i) z[i] = 0.0f;
    return z;
}

__device__ __forceinline__ v8f wmma_bf16(const FragBF& a, const FragBF& b, v8f c) {
    return __builtin_amdgcn_wmma_f32_16x16x32_bf16(false, a.v, false, b.v,
                                                   (short)0, c, false, false);
}

#if USE_TDM
typedef __attribute__((ext_vector_type(4))) unsigned int u32x4;
typedef __attribute__((ext_vector_type(8))) int          i32x8;
typedef __attribute__((ext_vector_type(4))) int          i32x4;

// TDM load of a 2-D bf16 tile: 32 rows x 64 elements, row stride 1024
// elements, into contiguous LDS at lds_off. One issue per wave; tracked
// with TENSORcnt. Descriptor layout per CDNA5 ISA ch.10.8 (D# groups 0/1).
__device__ __forceinline__ void tdm_load_tile_32x64(const unsigned short* gptr,
                                                    unsigned lds_off) {
    unsigned long long ga = (unsigned long long)gptr;
    u32x4 g0;
    g0[0] = 1u;                                           // count=1, user mode
    g0[1] = lds_off;                                      // lds_addr (bytes)
    g0[2] = (unsigned)(ga & 0xFFFFFFFFull);               // global_addr[31:0]
    g0[3] = (unsigned)((ga >> 32) & 0x1FFFFFFull)         // global_addr[56:32]
            | (2u << 30);                                 // type=2 ("image")
    i32x8 g1;
    g1[0] = (int)(1u << 16);        // workgroup_mask=0, data_size=1 (2 bytes)
    g1[1] = (int)(64u << 16);       // atomic_barrier=0 | tensor_dim0[15:0]=64
    g1[2] = (int)(32u << 16);       // tensor_dim0[31:16]=0 | tensor_dim1[15:0]=32
    g1[3] = (int)(64u << 16);       // tensor_dim1[31:16]=0 | tile_dim0=64
    g1[4] = (int)32u;               // tile_dim1=32 | tile_dim2=0
    g1[5] = (int)1024u;             // tensor_dim0_stride[31:0]=1024
    g1[6] = 0;                      // stride hi | tensor_dim1_stride lo
    g1[7] = 0;
    i32x4 z4 = {0, 0, 0, 0};
#if __clang_major__ >= 23
    i32x8 z8 = {0, 0, 0, 0, 0, 0, 0, 0};
    __builtin_amdgcn_tensor_load_to_lds(g0, g1, z4, z4, z8, 0);
#else
    __builtin_amdgcn_tensor_load_to_lds(g0, g1, z4, z4, 0);
#endif
}
#endif // USE_TDM

// ---------------------------------------------------------------------------
// fp32 -> bf16 conversion
// ---------------------------------------------------------------------------
__global__ void cvt_bf16_kernel(const float* __restrict__ src,
                                unsigned short* __restrict__ dst, int n) {
    int i = blockIdx.x * blockDim.x + threadIdx.x;
    if (i < n) dst[i] = f32_to_bf16(src[i]);
}

// ---------------------------------------------------------------------------
// Tiled bf16 GEMM with fp32 accumulation: C[M,N] = A[M,K] * B[K,N]
// 256 threads = 8 waves; block tile 64x64; wave tile 16x32; K-step 32.
// Writes fp32 (Cf) or bf16 (Cb), whichever pointer is non-null.
// ---------------------------------------------------------------------------
__global__ __launch_bounds__(256)
void gemm_bf16_kernel(const unsigned short* __restrict__ A,
                      const unsigned short* __restrict__ Bw,
                      float* __restrict__ Cf,
                      unsigned short* __restrict__ Cb,
                      int M, int N, int K) {
    __shared__ unsigned int   sA[64][16];   // [m][k/2] uint-packed bf16 pairs
    __shared__ unsigned short sB[64][36];   // [n][k], padded row stride

    const int tid  = threadIdx.x;
    const int lane = tid & 31;
    const int wv   = tid >> 5;          // 0..7
    const int half = lane >> 4;
    const int ln   = lane & 15;
    const int m0   = blockIdx.y * 64;
    const int n0   = blockIdx.x * 64;
    const int wm   = (wv & 3) * 16;     // wave M offset in block tile
    const int wn   = (wv >> 2) * 32;    // wave N offset in block tile

    v8f c0 = v8f_zero(), c1 = v8f_zero();

    for (int k0 = 0; k0 < K; k0 += 32) {
        // A tile: 64x32 bf16 = 1024 uints, 4 per thread, coalesced.
#pragma unroll
        for (int i = 0; i < 4; ++i) {
            int idx = tid + 256 * i;
            int r = idx >> 4, cu = idx & 15;
            sA[r][cu] = *(const unsigned int*)(A + (size_t)(m0 + r) * K + k0 + cu * 2);
        }
        // B tile: 32x64 bf16, transposed into [n][k].
#pragma unroll
        for (int i = 0; i < 4; ++i) {
            int idx = tid + 256 * i;
            int kr = idx >> 5, nu = idx & 31;
            unsigned int w = *(const unsigned int*)(Bw + (size_t)(k0 + kr) * N + n0 + nu * 2);
            sB[nu * 2][kr]     = (unsigned short)(w & 0xFFFFu);
            sB[nu * 2 + 1][kr] = (unsigned short)(w >> 16);
        }
        // Prefetch next K-tile into cache (global_prefetch_b8).
        if (k0 + 32 < K) {
            __builtin_prefetch(A + (size_t)(m0 + (tid & 63)) * K + k0 + 32, 0, 0);
            __builtin_prefetch(Bw + (size_t)(k0 + 32 + (tid & 31)) * N + n0, 0, 0);
        }
        __syncthreads();

        FragBF a, b0, b1;
#pragma unroll
        for (int v = 0; v < 8; ++v) {
            int kk = (v >> 2) * 16 + half * 8 + (v & 3) * 2;
            a.u[v]  = sA[wm + ln][kk >> 1];
            b0.u[v] = *(const unsigned int*)&sB[wn + ln][kk];
            b1.u[v] = *(const unsigned int*)&sB[wn + 16 + ln][kk];
        }
        c0 = wmma_bf16(a, b0, c0);
        c1 = wmma_bf16(a, b1, c1);
        __syncthreads();
    }

#pragma unroll
    for (int r = 0; r < 8; ++r) {
        int row  = m0 + wm + r + 8 * half;
        int col  = n0 + wn + ln;
        if (Cf) {
            Cf[(size_t)row * N + col]      = c0[r];
            Cf[(size_t)row * N + col + 16] = c1[r];
        } else {
            Cb[(size_t)row * N + col]      = f32_to_bf16(c0[r]);
            Cb[(size_t)row * N + col + 16] = f32_to_bf16(c1[r]);
        }
    }
}

// ---------------------------------------------------------------------------
// RoPE (fp32 in, bf16 out). One thread per (b,s,h,j), j in [0,32).
// ---------------------------------------------------------------------------
__global__ void rope_kernel(const float* __restrict__ src,
                            unsigned short* __restrict__ dst, float scale) {
    int idx = blockIdx.x * blockDim.x + threadIdx.x;
    if (idx >= BB * SS * HH * 32) return;
    int j = idx & 31;
    int h = (idx >> 5) & (HH - 1);
    int s = (idx >> 9) & (SS - 1);
    int b = idx >> 20;
    size_t base = (((size_t)b * SS + s) * HH + h) * DHD;
    float first  = src[base + j];
    float second = src[base + 32 + j];
    // timescale = 10000^(j/32); inv = exp(-j/32 * ln(10000))
    float inv_ts = __expf(-(float)j * (1.0f / 32.0f) * 9.210340371976184f);
    float ang = (float)s * inv_ts;
    float sn, cs;
    sincosf(ang, &sn, &cs);
    dst[base + j]      = f32_to_bf16((first * cs - second * sn) * scale);
    dst[base + 32 + j] = f32_to_bf16((second * cs + first * sn) * scale);
}

// ---------------------------------------------------------------------------
// Causal flash attention. 64 threads = 2 waves; each wave owns 16 q rows
// (Q kept in registers as two A-fragments) and streams 32-row K/V tiles
// through double-buffered per-wave LDS slices. On gfx1250 the tiles are
// fetched by the Tensor Data Mover (tensor_load_to_lds + s_wait_tensorcnt)
// with one-tile lookahead; fallback is a manual global->LDS copy.
// Online softmax in fp32; QK^T and PV in bf16 WMMA with fp32 accumulation.
// No block barriers (each wave uses only its own LDS slices).
// ---------------------------------------------------------------------------
__global__ __launch_bounds__(64)
void attn_kernel(const unsigned short* __restrict__ Qb,
                 const unsigned short* __restrict__ Kb,
                 const unsigned short* __restrict__ Vb,
                 unsigned short* __restrict__ Ob) {
    __shared__ unsigned short sK[2][2][32 * 64];  // [wave][buf][krow*64+dim]
    __shared__ unsigned short sV[2][2][32 * 64];
    __shared__ unsigned short sP[2][16 * 32];     // [wave][qrow*32+kcol]

    const int tid  = threadIdx.x;
    const int lane = tid & 31;
    const int w    = tid >> 5;          // 0..1
    const int half = lane >> 4;
    const int ln   = lane & 15;
    const int b    = blockIdx.z;
    const int h    = blockIdx.y;
    const int q0   = blockIdx.x * 32 + w * 16;

    // Per-(b,h) base pointers: row s lives at base + s*HH*DHD.
    const unsigned short* kbase = Kb + ((size_t)b * SS * HH + h) * DHD;
    const unsigned short* vbase = Vb + ((size_t)b * SS * HH + h) * DHD;

    // Q fragments (16 rows x 64 dims -> two 16x32 A-fragments), direct global.
    FragBF qa0, qa1;
    {
        const unsigned short* qrow = Qb + (((size_t)b * SS + q0 + ln) * HH + h) * DHD;
#pragma unroll
        for (int v = 0; v < 8; ++v) {
            int kk = (v >> 2) * 16 + half * 8 + (v & 3) * 2;
            qa0.u[v] = *(const unsigned int*)(qrow + kk);
            qa1.u[v] = *(const unsigned int*)(qrow + 32 + kk);
        }
    }

    float mstate[8], lstate[8];
    v8f acc[4];
#pragma unroll
    for (int r = 0; r < 8; ++r) { mstate[r] = -1e30f; lstate[r] = 0.0f; }
#pragma unroll
    for (int j = 0; j < 4; ++j) acc[j] = v8f_zero();

    const int ktiles = (q0 + 16 + 31) >> 5;

#if USE_TDM
    unsigned offK[2], offV[2];
#pragma unroll
    for (int i = 0; i < 2; ++i) {
        offK[i] = (unsigned)(unsigned long long)(&sK[w][i][0]);
        offV[i] = (unsigned)(unsigned long long)(&sV[w][i][0]);
    }
    // Prologue: fetch tile 0 into buffer 0.
    tdm_load_tile_32x64(kbase, offK[0]);
    tdm_load_tile_32x64(vbase, offV[0]);
#endif

    for (int kt = 0; kt < ktiles; ++kt) {
        const int kb  = kt * 32;
        const int buf = kt & 1;

#if USE_TDM
        if (kt + 1 < ktiles) {
            // Issue next tile into the other buffer, then wait for current
            // tile (2 oldest of 4 outstanding TDM ops; TDM is in-order).
            tdm_load_tile_32x64(kbase + (size_t)(kb + 32) * (HH * DHD), offK[1 - buf]);
            tdm_load_tile_32x64(vbase + (size_t)(kb + 32) * (HH * DHD), offV[1 - buf]);
            __builtin_amdgcn_s_wait_tensorcnt(2);
        } else {
            __builtin_amdgcn_s_wait_tensorcnt(0);
        }
#else
        // Manual copy: 32 rows x 64 bf16, one row per lane, 128B per row.
        {
            const uint4* gk = (const uint4*)(kbase + (size_t)(kb + lane) * (HH * DHD));
            const uint4* gv = (const uint4*)(vbase + (size_t)(kb + lane) * (HH * DHD));
            uint4* dk = (uint4*)&sK[w][buf][lane * 64];
            uint4* dv = (uint4*)&sV[w][buf][lane * 64];
#pragma unroll
            for (int i = 0; i < 8; ++i) { dk[i] = gk[i]; dv[i] = gv[i]; }
        }
        asm volatile("s_wait_dscnt 0x0" ::: "memory");
#endif

        // Scores: two 16x16 subtiles (k cols kb..kb+15 and kb+16..kb+31).
        v8f sc[2];
#pragma unroll
        for (int t = 0; t < 2; ++t) {
            FragBF f0, f1;
#pragma unroll
            for (int v = 0; v < 8; ++v) {
                int kk   = (v >> 2) * 16 + half * 8 + (v & 3) * 2;
                int rofs = (t * 16 + ln) * 64;   // B column = K row (row-major tile)
                f0.u[v] = *(const unsigned int*)&sK[w][buf][rofs + kk];
                f1.u[v] = *(const unsigned int*)&sK[w][buf][rofs + 32 + kk];
            }
            v8f s = v8f_zero();
            s = wmma_bf16(qa0, f0, s);
            s = wmma_bf16(qa1, f1, s);
            sc[t] = s;
        }

        // Online softmax. Row r+8*half, col = lane&15 within each subtile.
#pragma unroll
        for (int r = 0; r < 8; ++r) {
            int qi = q0 + r + 8 * half;
            float x0 = (kb + ln      <= qi) ? sc[0][r] : -1e30f;
            float x1 = (kb + 16 + ln <= qi) ? sc[1][r] : -1e30f;
            float tm = fmaxf(x0, x1);
            tm = fmaxf(tm, __shfl_xor(tm, 1, 32));
            tm = fmaxf(tm, __shfl_xor(tm, 2, 32));
            tm = fmaxf(tm, __shfl_xor(tm, 4, 32));
            tm = fmaxf(tm, __shfl_xor(tm, 8, 32));
            float nm   = fmaxf(mstate[r], tm);
            float corr = __expf(mstate[r] - nm);
            float p0   = __expf(x0 - nm);
            float p1   = __expf(x1 - nm);
            float rs = p0 + p1;
            rs += __shfl_xor(rs, 1, 32);
            rs += __shfl_xor(rs, 2, 32);
            rs += __shfl_xor(rs, 4, 32);
            rs += __shfl_xor(rs, 8, 32);
            lstate[r] = lstate[r] * corr + rs;
            mstate[r] = nm;
#pragma unroll
            for (int j = 0; j < 4; ++j) acc[j][r] = acc[j][r] * corr;
            int prow = (r + 8 * half) * 32;
            sP[w][prow + ln]      = f32_to_bf16(p0);
            sP[w][prow + 16 + ln] = f32_to_bf16(p1);
        }
        asm volatile("s_wait_dscnt 0x0" ::: "memory");

        // P (16x32) A-fragment, then PV into 4 accumulators (Dh = 4 x 16).
        FragBF pa;
#pragma unroll
        for (int v = 0; v < 8; ++v) {
            int kk = (v >> 2) * 16 + half * 8 + (v & 3) * 2;
            pa.u[v] = *(const unsigned int*)&sP[w][ln * 32 + kk];
        }
#pragma unroll
        for (int j = 0; j < 4; ++j) {
            FragBF vb;
#pragma unroll
            for (int v = 0; v < 8; ++v) {
                int kk = (v >> 2) * 16 + half * 8 + (v & 3) * 2;
                unsigned int lo = sV[w][buf][kk * 64 + j * 16 + ln];
                unsigned int hi = sV[w][buf][(kk + 1) * 64 + j * 16 + ln];
                vb.u[v] = lo | (hi << 16);
            }
            acc[j] = wmma_bf16(pa, vb, acc[j]);
        }
    }

    // Normalize and write O (bf16, [B,S,H,Dh] flat = [8192,1024]).
#pragma unroll
    for (int j = 0; j < 4; ++j) {
#pragma unroll
        for (int r = 0; r < 8; ++r) {
            int row = q0 + r + 8 * half;
            float val = acc[j][r] / lstate[r];
            Ob[(((size_t)b * SS + row) * HH + h) * DHD + j * 16 + ln] = f32_to_bf16(val);
        }
    }
}

// ---------------------------------------------------------------------------
// Host launcher
// ---------------------------------------------------------------------------
extern "C" void kernel_launch(void* const* d_in, const int* in_sizes, int n_in,
                              void* d_out, int out_size, void* d_ws, size_t ws_size,
                              hipStream_t stream) {
    (void)in_sizes; (void)n_in; (void)out_size; (void)ws_size;
    const float* x  = (const float*)d_in[0];
    const float* wq = (const float*)d_in[1];
    const float* wk = (const float*)d_in[2];
    const float* wv = (const float*)d_in[3];
    const float* wo = (const float*)d_in[4];
    float* out = (float*)d_out;

    char* ws = (char*)d_ws;
    const size_t MiB = (size_t)1 << 20;
    unsigned short* Xb  = (unsigned short*)(ws + 0);          // 16 MiB
    unsigned short* Wqb = (unsigned short*)(ws + 16 * MiB);   //  2 MiB
    unsigned short* Wkb = (unsigned short*)(ws + 18 * MiB);
    unsigned short* Wvb = (unsigned short*)(ws + 20 * MiB);
    unsigned short* Wob = (unsigned short*)(ws + 22 * MiB);
    float*          Qf  = (float*)(ws + 24 * MiB);            // 32 MiB
    float*          Kf  = (float*)(ws + 56 * MiB);            // 32 MiB
    unsigned short* Qh  = (unsigned short*)(ws + 88 * MiB);   // 16 MiB
    unsigned short* Kh  = (unsigned short*)(ws + 104 * MiB);
    unsigned short* Vh  = (unsigned short*)(ws + 120 * MiB);
    unsigned short* Oh  = (unsigned short*)(ws + 136 * MiB);  // ends 152 MiB

    const int nX = MROWS * DD;     // 8,388,608
    const int nW = DD * DD;        // 1,048,576
    cvt_bf16_kernel<<<(nX + 255) / 256, 256, 0, stream>>>(x,  Xb,  nX);
    cvt_bf16_kernel<<<(nW + 255) / 256, 256, 0, stream>>>(wq, Wqb, nW);
    cvt_bf16_kernel<<<(nW + 255) / 256, 256, 0, stream>>>(wk, Wkb, nW);
    cvt_bf16_kernel<<<(nW + 255) / 256, 256, 0, stream>>>(wv, Wvb, nW);
    cvt_bf16_kernel<<<(nW + 255) / 256, 256, 0, stream>>>(wo, Wob, nW);

    dim3 gg(DD / 64, MROWS / 64);  // (16, 128)
    gemm_bf16_kernel<<<gg, 256, 0, stream>>>(Xb, Wqb, Qf, nullptr, MROWS, DD, DD);
    gemm_bf16_kernel<<<gg, 256, 0, stream>>>(Xb, Wkb, Kf, nullptr, MROWS, DD, DD);
    gemm_bf16_kernel<<<gg, 256, 0, stream>>>(Xb, Wvb, nullptr, Vh, MROWS, DD, DD);

    const int nR = BB * SS * HH * 32;  // 4,194,304
    rope_kernel<<<(nR + 255) / 256, 256, 0, stream>>>(Qf, Qh, 0.125f); // 1/sqrt(64)
    rope_kernel<<<(nR + 255) / 256, 256, 0, stream>>>(Kf, Kh, 1.0f);

    attn_kernel<<<dim3(SS / 32, HH, BB), 64, 0, stream>>>(Qh, Kh, Vh, Oh);

    gemm_bf16_kernel<<<gg, 256, 0, stream>>>(Oh, Wob, out, nullptr, MROWS, DD, DD);
}
